// DualSceneAligner_11175504904452
// MI455X (gfx1250) — matrix operand order repaired and checked by reading, not compiled, not run on hardware
//
#include <hip/hip_runtime.h>
#include <hip/hip_bf16.h>
#include <math.h>

// ---------------------------------------------------------------------------
// DualSceneAligner on gfx1250 (CDNA5, wave32, WMMA).
// Dense GEMMs use v_wmma_f32_16x16x32_bf16. Weights are pre-converted once
// per launch to transposed, K-padded bf16 so LDS staging and fragment loads
// are fully vectorized (global_load_b64 / ds_load_b128).
// Graph attention (segment softmax + scatter-add) uses ordered-uint float
// atomicMax + float atomicAdd.
// ---------------------------------------------------------------------------

typedef __bf16 bf16_t;
typedef bf16_t v16bf __attribute__((ext_vector_type(16)));
typedef bf16_t v8bf  __attribute__((ext_vector_type(8)));
typedef bf16_t v2bf  __attribute__((ext_vector_type(2)));
typedef float  v8f   __attribute__((ext_vector_type(8)));

#define NNODES  50000
#define EGEOM   800000
#define ETEXT   400000
#define NGRAPHS 64
#define HIDDEN  128
#define NHEAD   2
#define DHEAD   64

// ------------------------------ device helpers -----------------------------

__device__ __forceinline__ unsigned fkey(float f) {
  unsigned u = __float_as_uint(f);
  return (u & 0x80000000u) ? ~u : (u | 0x80000000u);
}
__device__ __forceinline__ float funkey(unsigned k) {
  unsigned u = (k & 0x80000000u) ? (k & 0x7FFFFFFFu) : ~k;
  return __uint_as_float(u);
}
__device__ __forceinline__ float gelu_f(float x) {
  return 0.5f * x * (1.0f + erff(x * 0.70710678118654752f));
}

// ------------------------------ utility kernels ----------------------------

__global__ void fill_f32_k(float* p, float v, long n) {
  long i = (long)blockIdx.x * blockDim.x + threadIdx.x;
  if (i < n) p[i] = v;
}
__global__ void fill_key_k(unsigned* p, float v, long n) {
  long i = (long)blockIdx.x * blockDim.x + threadIdx.x;
  if (i < n) p[i] = fkey(v);
}
// weight convert: f32 [K,N] row-major -> bf16 transposed [N,Kpad], zero-padded K
__global__ void cvt_bf16_T_k(const float* __restrict__ w, bf16_t* __restrict__ wt,
                             int K, int N, int Kpad) {
  long i = (long)blockIdx.x * blockDim.x + threadIdx.x;
  if (i >= (long)N * Kpad) return;
  int n = (int)(i / Kpad), k = (int)(i % Kpad);
  wt[i] = (k < K) ? (bf16_t)w[(long)k * N + n] : (bf16_t)0.0f;
}

// ------------------------------ WMMA GEMM ----------------------------------
// C[M,N] = act( A[M,K](f32) @ W[K,N] + bias ),  W given as bf16 Wt[N,Kpad].
// Block: 256 threads = 8 waves; tile 64(M) x 64(N); K stepped by 32.
// Wave (mt,nt) owns a 16x32 D tile -> two v_wmma_f32_16x16x32_bf16 per step.
//
// LDS layouts are fragment-ordered so each lane reads its whole fragment as
// one contiguous 32B chunk:
//   A: k-groups swizzled [0-7,16-23,8-15,24-31]  -> af  = As2[row][half*16..]
//   B: transposed Bst[col][p], same k swizzle as A             -> bfv = Bst[col][half*16..]

__global__ __launch_bounds__(256)
void gemm_bf16_k(const float* __restrict__ A, int lda,
                 const bf16_t* __restrict__ Wt, int Kpad,
                 const float* __restrict__ bias,
                 float* __restrict__ C,
                 int M, int K, int N, int act) {
  __shared__ __align__(32) bf16_t As2[64][32];
  __shared__ __align__(32) bf16_t Bst[64][32];
  const int tid  = threadIdx.x;
  const int wid  = tid >> 5;        // wave id 0..7 (wave32)
  const int lane = tid & 31;
  const int half = lane >> 4;       // lane-half 0/1
  const int lr   = lane & 15;
  const int rowBase = blockIdx.x * 64;
  const int colBase = blockIdx.y * 64;
  const int mt = wid & 3;           // 16-row sub-tile
  const int nt = wid >> 2;          // 32-col sub-tile (two 16-col WMMA tiles)
  const bool fullM = (rowBase + 64 <= M);

  v8f acc0 = {}, acc1 = {};
  for (int k0 = 0; k0 < K; k0 += 32) {
    // ---- stage A tile (f32 -> bf16), fragment-ordered k swizzle ----
    if (fullM && (k0 + 32 <= K)) {
      // fast path: vector loads, no guards (64 rows x 16 float2 chunks)
      for (int i = tid; i < 64 * 16; i += 256) {
        int r = i >> 4, kp = (i & 15) * 2;
        float2 v = *(const float2*)(A + (long)(rowBase + r) * lda + k0 + kp);
        int p = (kp & 7) | ((kp & 16) >> 1) | ((kp & 8) << 1);
        v2bf pk = {(bf16_t)v.x, (bf16_t)v.y};
        *(v2bf*)&As2[r][p] = pk;
      }
    } else {
      // edge path (last M block / K remainder): guarded scalar
      for (int i = tid; i < 64 * 32; i += 256) {
        int r = i >> 5, k = i & 31;
        int gr = rowBase + r, gk = k0 + k;
        float v = (gr < M && gk < K) ? A[(long)gr * lda + gk] : 0.0f;
        int p = (k & 7) | ((k & 16) >> 1) | ((k & 8) << 1);
        As2[r][p] = (bf16_t)v;
      }
    }
    // ---- stage B tile from transposed bf16 weights (contiguous in k) ----
    {
      int c = tid >> 2, kc = (tid & 3) * 8;   // 64 cols x 4 chunks = 256 threads
      v8bf v = {};
      if (colBase + c < N)
        v = *(const v8bf*)(Wt + (long)(colBase + c) * Kpad + k0 + kc);
      *(v8bf*)&Bst[c][((kc & 8) << 1) | ((kc & 16) >> 1)] = v;
    }
    __syncthreads();

    const int arow = mt * 16 + lr;
    v16bf af = *(const v16bf*)&As2[arow][half * 16];
    v16bf b0 = *(const v16bf*)&Bst[nt * 32 + lr][half * 16];
    v16bf b1 = *(const v16bf*)&Bst[nt * 32 + 16 + lr][half * 16];
    acc0 = __builtin_amdgcn_wmma_f32_16x16x32_bf16(false, af, false, b0,
                                                   (short)0, acc0, false, false);
    acc1 = __builtin_amdgcn_wmma_f32_16x16x32_bf16(false, af, false, b1,
                                                   (short)0, acc1, false, false);
    __syncthreads();
  }

  const int col0 = colBase + nt * 32 + lr;
  const int col1 = col0 + 16;
  const int rbase = rowBase + mt * 16 + half * 8;
  const float bv0 = (bias && col0 < N) ? bias[col0] : 0.0f;
  const float bv1 = (bias && col1 < N) ? bias[col1] : 0.0f;
#pragma unroll
  for (int v = 0; v < 8; ++v) {
    int row = rbase + v;
    if (row < M) {
      if (col0 < N) {
        float o = acc0[v] + bv0;
        if (act == 1)      o = gelu_f(o);
        else if (act == 2) o = 1.0f / (1.0f + expf(-o));
        C[(long)row * N + col0] = o;
      }
      if (col1 < N) {
        float o = acc1[v] + bv1;
        if (act == 1)      o = gelu_f(o);
        else if (act == 2) o = 1.0f / (1.0f + expf(-o));
        C[(long)row * N + col1] = o;
      }
    }
  }
}

// ------------------------------ LayerNorm ----------------------------------
// One wave per row (wave32 shuffle reductions). Optional fused GELU.

__global__ __launch_bounds__(256)
void layernorm_k(const float* __restrict__ in, float* __restrict__ out,
                 const float* __restrict__ g, const float* __restrict__ b,
                 int N, int C, int gelu_after) {
  int wid = threadIdx.x >> 5, lane = threadIdx.x & 31;
  int row = blockIdx.x * 8 + wid;
  if (row >= N) return;
  const float* x = in + (long)row * C;
  float s = 0.0f;
  for (int c = lane; c < C; c += 32) s += x[c];
#pragma unroll
  for (int m = 16; m; m >>= 1) s += __shfl_xor(s, m, 32);
  float mean = s / (float)C;
  float v = 0.0f;
  for (int c = lane; c < C; c += 32) { float d = x[c] - mean; v += d * d; }
#pragma unroll
  for (int m = 16; m; m >>= 1) v += __shfl_xor(v, m, 32);
  float inv = rsqrtf(v / (float)C + 1e-5f);
  float* o = out + (long)row * C;
  for (int c = lane; c < C; c += 32) {
    float y = (x[c] - mean) * inv * g[c] + b[c];
    if (gelu_after) y = gelu_f(y);
    o[c] = y;
  }
}

// ------------------------------ GAT kernels --------------------------------

// per-node attention scores: ss[n,h] = h[n,h,:].a_s[h]  (and a_d)
__global__ void node_score_k(const float* __restrict__ h,
                             const float* __restrict__ a_s,
                             const float* __restrict__ a_d,
                             float* __restrict__ ss, float* __restrict__ ds, int N) {
  int t = blockIdx.x * blockDim.x + threadIdx.x;
  if (t >= N * NHEAD) return;
  int n = t >> 1, hd = t & 1;
  const float* hp = h + (long)n * HIDDEN + hd * DHEAD;
  const float* as = a_s + hd * DHEAD;
  const float* ad = a_d + hd * DHEAD;
  float s = 0.0f, d = 0.0f;
#pragma unroll 8
  for (int i = 0; i < DHEAD; ++i) { s += hp[i] * as[i]; d += hp[i] * ad[i]; }
  ss[t] = s; ds[t] = d;
}

// tiny precompute: aeproj[j,h] = sum_d We[j, h*64+d] * a_e[h,d]
__global__ void aeproj_k(const float* __restrict__ We, const float* __restrict__ a_e,
                         float* __restrict__ out, int edim) {
  int t = blockIdx.x * blockDim.x + threadIdx.x;
  if (t >= edim * NHEAD) return;
  int j = t >> 1, hd = t & 1;
  float s = 0.0f;
  for (int d = 0; d < DHEAD; ++d) s += We[j * HIDDEN + hd * DHEAD + d] * a_e[hd * DHEAD + d];
  out[t] = s;
}
// text relation tables: eproj[r,c] = sum_k rel[r,k] * We[k,c]   (10 x 128)
__global__ void relproj_k(const float* __restrict__ rel, const float* __restrict__ We,
                          float* __restrict__ table) {
  int t = blockIdx.x * blockDim.x + threadIdx.x;
  if (t >= 10 * HIDDEN) return;
  int r = t >> 7, c = t & 127;
  float s = 0.0f;
  for (int k = 0; k < 64; ++k) s += rel[r * 64 + k] * We[k * HIDDEN + c];
  table[t] = s;
}
__global__ void esctab_k(const float* __restrict__ table, const float* __restrict__ a_e,
                         float* __restrict__ esc) {
  int t = blockIdx.x * blockDim.x + threadIdx.x;
  if (t >= 10 * NHEAD) return;
  int r = t >> 1, hd = t & 1;
  float s = 0.0f;
  for (int d = 0; d < DHEAD; ++d) s += table[r * HIDDEN + hd * DHEAD + d] * a_e[hd * DHEAD + d];
  esc[t] = s;
}

// logits + segment max (ordered-uint float atomicMax)
__global__ void edge_logit_geom_k(const int* __restrict__ src, const int* __restrict__ dst,
                                  const float* __restrict__ eattr, const float* __restrict__ aeproj,
                                  const float* __restrict__ ss, const float* __restrict__ ds,
                                  float* __restrict__ logit, unsigned* __restrict__ mkey, int E) {
  int t = blockIdx.x * blockDim.x + threadIdx.x;
  if (t >= E * NHEAD) return;
  int e = t >> 1, hd = t & 1;
  int sN = src[e], dN = dst[e];
  float es = 0.0f;
#pragma unroll
  for (int j = 0; j < 8; ++j) es += eattr[(long)e * 8 + j] * aeproj[j * 2 + hd];
  float l = ss[sN * 2 + hd] + ds[dN * 2 + hd] + es;
  l = (l > 0.0f) ? l : 0.2f * l;   // leaky_relu 0.2
  logit[t] = l;
  atomicMax(&mkey[dN * 2 + hd], fkey(l));
}
__global__ void edge_logit_text_k(const int* __restrict__ src, const int* __restrict__ dst,
                                  const int* __restrict__ ttype, const float* __restrict__ esctab,
                                  const float* __restrict__ ss, const float* __restrict__ ds,
                                  float* __restrict__ logit, unsigned* __restrict__ mkey, int E) {
  int t = blockIdx.x * blockDim.x + threadIdx.x;
  if (t >= E * NHEAD) return;
  int e = t >> 1, hd = t & 1;
  int sN = src[e], dN = dst[e];
  float l = ss[sN * 2 + hd] + ds[dN * 2 + hd] + esctab[ttype[e] * 2 + hd];
  l = (l > 0.0f) ? l : 0.2f * l;
  logit[t] = l;
  atomicMax(&mkey[dN * 2 + hd], fkey(l));
}

// w = exp(logit - m[dst]); denom[dst] += w   (in-place on logit buffer)
__global__ void edge_expsum_k(const int* __restrict__ dst, const unsigned* __restrict__ mkey,
                              float* __restrict__ wexp, float* __restrict__ denom, int E) {
  int t = blockIdx.x * blockDim.x + threadIdx.x;
  if (t >= E * NHEAD) return;
  int e = t >> 1, hd = t & 1;
  int dN = dst[e];
  float w = expf(wexp[t] - funkey(mkey[dN * 2 + hd]));
  wexp[t] = w;
  atomicAdd(&denom[dN * 2 + hd], w);
}

// message: aggr[dst,c] += alpha * (h[src,c] + e[c]); one thread per (edge, channel)
__global__ void edge_msg_geom_k(const int* __restrict__ src, const int* __restrict__ dst,
                                const float* __restrict__ eattr, const float* __restrict__ We,
                                const float* __restrict__ h, const float* __restrict__ wexp,
                                const float* __restrict__ denom, float* __restrict__ aggr, int E) {
  long t = (long)blockIdx.x * blockDim.x + threadIdx.x;
  if (t >= (long)E * HIDDEN) return;
  int e = (int)(t >> 7), c = (int)(t & 127), hd = c >> 6;
  int dN = dst[e];
  float alpha = wexp[e * 2 + hd] / (denom[dN * 2 + hd] + 1e-16f);
  float ev = 0.0f;
#pragma unroll
  for (int j = 0; j < 8; ++j) ev += eattr[(long)e * 8 + j] * We[j * HIDDEN + c];
  int sN = src[e];
  atomicAdd(&aggr[(long)dN * HIDDEN + c], alpha * (h[(long)sN * HIDDEN + c] + ev));
}
__global__ void edge_msg_text_k(const int* __restrict__ src, const int* __restrict__ dst,
                                const int* __restrict__ ttype, const float* __restrict__ eproj,
                                const float* __restrict__ h, const float* __restrict__ wexp,
                                const float* __restrict__ denom, float* __restrict__ aggr, int E) {
  long t = (long)blockIdx.x * blockDim.x + threadIdx.x;
  if (t >= (long)E * HIDDEN) return;
  int e = (int)(t >> 7), c = (int)(t & 127), hd = c >> 6;
  int dN = dst[e];
  float alpha = wexp[e * 2 + hd] / (denom[dN * 2 + hd] + 1e-16f);
  float ev = eproj[ttype[e] * HIDDEN + c];
  int sN = src[e];
  atomicAdd(&aggr[(long)dN * HIDDEN + c], alpha * (h[(long)sN * HIDDEN + c] + ev));
}

// out = aggr + bias, optional ELU
__global__ void bias_act_k(const float* __restrict__ in, const float* __restrict__ b,
                           float* __restrict__ out, long n, int C, int elu) {
  long t = (long)blockIdx.x * blockDim.x + threadIdx.x;
  if (t >= n) return;
  float x = in[t] + b[(int)(t % C)];
  if (elu) x = (x > 0.0f) ? x : expm1f(x);
  out[t] = x;
}

// ------------------------------ fusion / pooling ---------------------------

__global__ void concat_k(const float* __restrict__ g, const float* __restrict__ t,
                         float* __restrict__ comb, int N) {
  long i = (long)blockIdx.x * blockDim.x + threadIdx.x;
  if (i >= (long)N * 256) return;
  int n = (int)(i >> 8), c = (int)(i & 255);
  comb[i] = (c < 128) ? g[(long)n * 128 + c] : t[(long)n * 128 + (c - 128)];
}
__global__ void fuse_mix_k(const float* __restrict__ gate, const float* __restrict__ g,
                           const float* __restrict__ t, const float* __restrict__ f2,
                           float* __restrict__ out, long n) {
  long i = (long)blockIdx.x * blockDim.x + threadIdx.x;
  if (i >= n) return;
  float ga = gate[i];
  out[i] = ga * g[i] + (1.0f - ga) * t[i] + 0.1f * f2[i];
}
__global__ void pool_accum_k(const float* __restrict__ x, const int* __restrict__ bidx,
                             float* __restrict__ psum, float* __restrict__ cnt, int N) {
  long t = (long)blockIdx.x * blockDim.x + threadIdx.x;
  if (t >= (long)N * HIDDEN) return;
  int n = (int)(t >> 7), c = (int)(t & 127);
  int b = bidx[n];
  atomicAdd(&psum[(long)b * HIDDEN + c], x[t]);
  if (c == 0) atomicAdd(&cnt[b], 1.0f);
}
__global__ void pool_div_k(float* __restrict__ psum, const float* __restrict__ cnt, int B) {
  int t = blockIdx.x * blockDim.x + threadIdx.x;
  if (t >= B * HIDDEN) return;
  psum[t] /= fmaxf(cnt[t >> 7], 1.0f);
}

// ------------------------------ host orchestration -------------------------

static inline unsigned nblk(long n, int bs) { return (unsigned)((n + bs - 1) / bs); }
static inline int kpad32(int K) { return (K + 31) & ~31; }

struct GatP { const float *W, *We, *a_d, *a_e, *a_s, *b; };

extern "C" void kernel_launch(void* const* d_in, const int* in_sizes, int n_in,
                              void* d_out, int out_size, void* d_ws, size_t ws_size,
                              hipStream_t stream) {
  (void)in_sizes; (void)n_in; (void)out_size; (void)ws_size;
  const int N = NNODES, Eg = EGEOM, Et = ETEXT;

  // ---- top-level inputs (setup_inputs insertion order) ----
  const float* nf_src   = (const float*)d_in[0];
  const int*   ge_src   = (const int*)d_in[1];   // [2, Eg]: src row then dst row
  const float* ga_src   = (const float*)d_in[2];
  const int*   te_src   = (const int*)d_in[3];
  const int*   ta_src   = (const int*)d_in[4];
  const int*   b_src    = (const int*)d_in[5];
  const float* nf_ref   = (const float*)d_in[6];
  const int*   ge_ref   = (const int*)d_in[7];
  const float* ga_ref   = (const float*)d_in[8];
  const int*   te_ref   = (const int*)d_in[9];
  const int*   ta_ref   = (const int*)d_in[10];
  const int*   b_ref    = (const int*)d_in[11];
  // d_in[12] = num_graphs scalar (fixed at 64 in this workload)

  // ---- params: JAX pytree leaves, dicts in sorted-key order ----
  const int PB = 13;
  auto P = [&](int i) { return (const float*)d_in[PB + i]; };
  const float *enc1_b = P(0),  *enc1_w = P(1);
  const float *enc2_b = P(2),  *enc2_w = P(3);
  const float *ln1_b  = P(4),  *ln1_g  = P(5);
  const float *ln2_b  = P(6),  *ln2_g  = P(7);
  const float *fin1_b = P(8),  *fin1_w = P(9);
  const float *fin2_b = P(10), *fin2_w = P(11);
  const float *finln_b = P(12), *finln_g = P(13);
  const float *fgate_b = P(14), *fgate_w = P(15);
  const float *fl1_b = P(16), *fl1_w = P(17);
  const float *fl2_b = P(18), *fl2_w = P(19);
  const float *fln_b = P(20), *fln_g = P(21);
  const float *fnorm_b = P(22), *fnorm_g = P(23);
  GatP gg[2], gt[2];
  for (int l = 0; l < 2; ++l) { int o = 24 + l * 6; gg[l] = {P(o), P(o+1), P(o+2), P(o+3), P(o+4), P(o+5)}; }
  for (int l = 0; l < 2; ++l) { int o = 36 + l * 6; gt[l] = {P(o), P(o+1), P(o+2), P(o+3), P(o+4), P(o+5)}; }
  const float *normg_b = P(48), *normg_g = P(49);
  const float *normt_b = P(50), *normt_g = P(51);
  const float *rel_emb = P(52);

  // ---- workspace bump allocator ----
  char* ws = (char*)d_ws;
  size_t off = 0;
  auto alloc = [&](size_t bytes) -> void* {
    off = (off + 255) & ~(size_t)255;
    void* p = ws + off;
    off += bytes;
    return p;
  };
  // transposed, K-padded bf16 weights
  const int KP518 = kpad32(518);   // 544
  bf16_t* wb_enc1 = (bf16_t*)alloc((size_t)256 * KP518 * 2);   // [N=256][Kp=544]
  bf16_t* wb_enc2 = (bf16_t*)alloc((size_t)128 * 256 * 2);     // [128][256]
  bf16_t* wb_ggW[2] = {(bf16_t*)alloc(128 * 128 * 2), (bf16_t*)alloc(128 * 128 * 2)};
  bf16_t* wb_gtW[2] = {(bf16_t*)alloc(128 * 128 * 2), (bf16_t*)alloc(128 * 128 * 2)};
  bf16_t* wb_fgate = (bf16_t*)alloc(128 * 256 * 2);
  bf16_t* wb_fl1   = (bf16_t*)alloc(128 * 256 * 2);
  bf16_t* wb_fl2   = (bf16_t*)alloc(128 * 128 * 2);
  bf16_t* wb_fin1  = (bf16_t*)alloc(256 * 128 * 2);
  bf16_t* wb_fin2  = (bf16_t*)alloc(256 * 256 * 2);
  // f32 activations
  float* x256   = (float*)alloc((size_t)N * 256 * 4);
  float* comb   = (float*)alloc((size_t)N * 256 * 4);
  float* xbuf   = (float*)alloc((size_t)N * 128 * 4);
  float* hbuf   = (float*)alloc((size_t)N * 128 * 4);
  float* aggr   = (float*)alloc((size_t)N * 128 * 4);
  float* l1out  = (float*)alloc((size_t)N * 128 * 4);
  float* gbuf   = (float*)alloc((size_t)N * 128 * 4);
  float* tbuf   = (float*)alloc((size_t)N * 128 * 4);
  float* gatev  = (float*)alloc((size_t)N * 128 * 4);
  float* f1     = (float*)alloc((size_t)N * 128 * 4);
  float* f2     = (float*)alloc((size_t)N * 128 * 4);
  float* fused  = (float*)alloc((size_t)N * 128 * 4);
  float* ss     = (float*)alloc((size_t)N * 2 * 4);
  float* ds     = (float*)alloc((size_t)N * 2 * 4);
  float* denom  = (float*)alloc((size_t)N * 2 * 4);
  unsigned* mkey = (unsigned*)alloc((size_t)N * 2 * 4);
  float* wexp   = (float*)alloc((size_t)Eg * 2 * 4);   // also holds logits in-place
  float* aeproj = (float*)alloc(16 * 4);
  float* eproj  = (float*)alloc(10 * 128 * 4);
  float* esctab = (float*)alloc(20 * 4);
  float* psum   = (float*)alloc((size_t)NGRAPHS * 128 * 4);
  float* cnt    = (float*)alloc((size_t)NGRAPHS * 4);
  float* y1     = (float*)alloc((size_t)NGRAPHS * 256 * 4);

  // ---- convert weights to transposed/padded bf16 (once per launch) ----
  auto CVT = [&](const float* s, bf16_t* d, int K, int Nc) {
    long n = (long)Nc * kpad32(K);
    cvt_bf16_T_k<<<nblk(n, 256), 256, 0, stream>>>(s, d, K, Nc, kpad32(K));
  };
  CVT(enc1_w, wb_enc1, 518, 256);
  CVT(enc2_w, wb_enc2, 256, 128);
  CVT(gg[0].W, wb_ggW[0], 128, 128);  CVT(gg[1].W, wb_ggW[1], 128, 128);
  CVT(gt[0].W, wb_gtW[0], 128, 128);  CVT(gt[1].W, wb_gtW[1], 128, 128);
  CVT(fgate_w, wb_fgate, 256, 128);
  CVT(fl1_w, wb_fl1, 256, 128);
  CVT(fl2_w, wb_fl2, 128, 128);
  CVT(fin1_w, wb_fin1, 128, 256);
  CVT(fin2_w, wb_fin2, 256, 256);

  auto GEMM = [&](const float* A, int lda, const bf16_t* Wt, const float* bias,
                  float* C, int M, int K, int Nc, int act) {
    dim3 grid(nblk(M, 64), nblk(Nc, 64));
    gemm_bf16_k<<<grid, 256, 0, stream>>>(A, lda, Wt, kpad32(K), bias, C, M, K, Nc, act);
  };
  auto LN = [&](const float* in, float* out, const float* g, const float* b,
                int rows, int C, int gelu) {
    layernorm_k<<<nblk(rows, 8), 256, 0, stream>>>(in, out, g, b, rows, C, gelu);
  };
  auto FILL = [&](float* p, float v, long n) {
    fill_f32_k<<<nblk(n, 256), 256, 0, stream>>>(p, v, n);
  };

  // ---- one edge-conditioned GAT layer ----
  auto run_gat = [&](const float* xin, const int* esrc, const int* edst, int E,
                     bool text, const float* eattr, const int* ttype,
                     const GatP& gp, const bf16_t* Wb, float* outBuf, int elu) {
    GEMM(xin, 128, Wb, nullptr, hbuf, N, 128, 128, 0);                 // h = x @ W
    node_score_k<<<nblk((long)N * 2, 256), 256, 0, stream>>>(hbuf, gp.a_s, gp.a_d, ss, ds, N);
    fill_key_k<<<nblk((long)N * 2, 256), 256, 0, stream>>>(mkey, -3.0e38f, (long)N * 2);
    FILL(denom, 0.0f, (long)N * 2);
    FILL(aggr, 0.0f, (long)N * 128);
    if (!text) {
      aeproj_k<<<1, 64, 0, stream>>>(gp.We, gp.a_e, aeproj, 8);
      edge_logit_geom_k<<<nblk((long)E * 2, 256), 256, 0, stream>>>(
          esrc, edst, eattr, aeproj, ss, ds, wexp, mkey, E);
    } else {
      relproj_k<<<5, 256, 0, stream>>>(rel_emb, gp.We, eproj);
      esctab_k<<<1, 32, 0, stream>>>(eproj, gp.a_e, esctab);
      edge_logit_text_k<<<nblk((long)E * 2, 256), 256, 0, stream>>>(
          esrc, edst, ttype, esctab, ss, ds, wexp, mkey, E);
    }
    edge_expsum_k<<<nblk((long)E * 2, 256), 256, 0, stream>>>(edst, mkey, wexp, denom, E);
    if (!text)
      edge_msg_geom_k<<<nblk((long)E * 128, 256), 256, 0, stream>>>(
          esrc, edst, eattr, gp.We, hbuf, wexp, denom, aggr, E);
    else
      edge_msg_text_k<<<nblk((long)E * 128, 256), 256, 0, stream>>>(
          esrc, edst, ttype, eproj, hbuf, wexp, denom, aggr, E);
    bias_act_k<<<nblk((long)N * 128, 256), 256, 0, stream>>>(
        aggr, gp.b, outBuf, (long)N * 128, 128, elu);
  };

  // ---- full scene encoder ----
  auto encode = [&](const float* nf, const int* ge, const float* ga,
                    const int* te, const int* ta, const int* bidx, float* outp) {
    const int *geS = ge, *geD = ge + Eg;
    const int *teS = te, *teD = te + Et;
    // node encoder
    GEMM(nf, 518, wb_enc1, enc1_b, x256, N, 518, 256, 0);
    LN(x256, x256, ln1_g, ln1_b, N, 256, 1);                            // LN + GELU
    GEMM(x256, 256, wb_enc2, enc2_b, xbuf, N, 256, 128, 0);
    LN(xbuf, xbuf, ln2_g, ln2_b, N, 128, 0);
    // geometric GAT stack
    run_gat(xbuf, geS, geD, Eg, false, ga, nullptr, gg[0], wb_ggW[0], l1out, 1);
    run_gat(l1out, geS, geD, Eg, false, ga, nullptr, gg[1], wb_ggW[1], gbuf, 0);
    LN(gbuf, gbuf, normg_g, normg_b, N, 128, 0);                        // g
    // textual GAT stack (input = g)
    run_gat(gbuf, teS, teD, Et, true, nullptr, ta, gt[0], wb_gtW[0], l1out, 1);
    run_gat(l1out, teS, teD, Et, true, nullptr, ta, gt[1], wb_gtW[1], tbuf, 0);
    LN(tbuf, tbuf, normt_g, normt_b, N, 128, 0);                        // t
    // fusion
    concat_k<<<nblk((long)N * 256, 256), 256, 0, stream>>>(gbuf, tbuf, comb, N);
    GEMM(comb, 256, wb_fgate, fgate_b, gatev, N, 256, 128, 2);          // sigmoid gate
    LN(comb, comb, fln_g, fln_b, N, 256, 0);                            // ln(comb)
    GEMM(comb, 256, wb_fl1, fl1_b, f1, N, 256, 128, 1);                 // gelu
    GEMM(f1, 128, wb_fl2, fl2_b, f2, N, 128, 128, 0);
    fuse_mix_k<<<nblk((long)N * 128, 256), 256, 0, stream>>>(gatev, gbuf, tbuf, f2, fused, (long)N * 128);
    LN(fused, fused, fnorm_g, fnorm_b, N, 128, 0);
    // mean pool per graph
    FILL(psum, 0.0f, (long)NGRAPHS * 128);
    FILL(cnt, 0.0f, NGRAPHS);
    pool_accum_k<<<nblk((long)N * 128, 256), 256, 0, stream>>>(fused, bidx, psum, cnt, N);
    pool_div_k<<<nblk((long)NGRAPHS * 128, 256), 256, 0, stream>>>(psum, cnt, NGRAPHS);
    // final head
    GEMM(psum, 128, wb_fin1, fin1_b, y1, NGRAPHS, 128, 256, 0);
    LN(y1, y1, finln_g, finln_b, NGRAPHS, 256, 1);                      // LN + GELU
    GEMM(y1, 256, wb_fin2, fin2_b, outp, NGRAPHS, 256, 256, 0);
  };

  float* out = (float*)d_out;
  encode(nf_src, ge_src, ga_src, te_src, ta_src, b_src, out);
  encode(nf_ref, ge_ref, ga_ref, te_ref, ta_ref, b_ref, out + (size_t)NGRAPHS * 256);
}
